// XMU_MultiHeadAttention_9457517986474
// MI455X (gfx1250) — matrix operand order repaired and checked by reading, not compiled, not run on hardware
//
#include <hip/hip_runtime.h>

// ---------------------------------------------------------------------------
// MI455X (gfx1250) multi-head attention, bf16 WMMA pipeline.
//
// Roofline: ~103 GFLOP total vs ~200MB of essential HBM traffic (~8us at
// 23.3 TB/s) -> matrix-core bound. All three GEMMs and both attention
// contractions run on v_wmma_f32_16x16x32_bf16 (f32 accumulate). Attention
// is flash-style: the B*H*S*S score matrix (512MB) never touches memory.
// GEMM tile staging uses CDNA5 async global->LDS DMA (ASYNCcnt) with double
// buffering when the toolchain exposes the builtins.
// ---------------------------------------------------------------------------

typedef unsigned short u16;
typedef __attribute__((ext_vector_type(16))) __bf16 bf16x16;
typedef __attribute__((ext_vector_type(8)))  float  v8f;
typedef int vint4 __attribute__((vector_size(16)));   // matches builtin param type

#if __has_builtin(__builtin_amdgcn_global_load_async_to_lds_b128) && \
    __has_builtin(__builtin_amdgcn_s_wait_asynccnt)
#define USE_ASYNC_COPY 1
#else
#define USE_ASYNC_COPY 0
#endif

#define GLOBAL_AS(p) ((__attribute__((address_space(1))) vint4*)(p))
#define LDS_AS(p)    ((__attribute__((address_space(3))) vint4*)(p))

union Frag {            // one 16x32 bf16 WMMA operand: 8 VGPRs / lane
    bf16x16 v;
    uint4   q[2];
};

// float -> bf16 (round to nearest even), pure integer math
__device__ __forceinline__ u16 f2bf(float f) {
    unsigned int u = __builtin_bit_cast(unsigned int, f);
    unsigned int r = 0x7FFFu + ((u >> 16) & 1u);
    return (u16)((u + r) >> 16);
}

// ---------------------------------------------------------------------------
// One-time converts
// ---------------------------------------------------------------------------
__global__ __launch_bounds__(256) void to_bf16_kernel(const float* __restrict__ in,
                                                      u16* __restrict__ out, int n) {
    int i = blockIdx.x * 256 + threadIdx.x;
    if (i < n) out[i] = f2bf(in[i]);
}

// in[R][C] f32 -> out[C][R] bf16 (so GEMM B-fragments read contiguous K)
__global__ __launch_bounds__(256) void transpose_to_bf16_kernel(const float* __restrict__ in,
                                                                u16* __restrict__ out,
                                                                int R, int C) {
    size_t idx = (size_t)blockIdx.x * 256 + threadIdx.x;
    if (idx >= (size_t)R * C) return;
    int r = (int)(idx % R);
    int c = (int)(idx / R);
    out[idx] = f2bf(in[(size_t)r * C + c]);
}

// ---------------------------------------------------------------------------
// bf16 GEMM:  C[M][N] = A[M][K] * Bt[N][K]^T + bias[N]
// Block tile 128x128, 8 waves (4 rows x 2 cols of 32x64 wave tiles),
// K-step 32 == WMMA depth. Double-buffered LDS tiles; staging uses
// global_load_async_to_lds_b128 (ASYNCcnt) when available so the next tile's
// DMA overlaps the current tile's WMMAs with one barrier per K-step.
// Fragments are two aligned ds_load_b128 matching the CDNA5 16-bit layout.
// ---------------------------------------------------------------------------
template <bool OUT_BF16>
__global__ __launch_bounds__(256) void gemm_bf16_kernel(const u16* __restrict__ A,
                                                        const u16* __restrict__ Bt,
                                                        const float* __restrict__ bias,
                                                        void* __restrict__ Cout,
                                                        int M, int N, int K) {
    __shared__ u16 As[2][128 * 32];
    __shared__ u16 Bs[2][128 * 32];

    const int tid   = threadIdx.x;
    const int lane  = tid & 31;
    const int wave  = tid >> 5;
    const int wm    = wave >> 1;        // 0..3  (M direction)
    const int wn    = wave & 1;         // 0..1  (N direction)
    const int lrow  = lane & 15;
    const int lhalf = lane >> 4;
    const int m0    = blockIdx.y * 128;
    const int n0    = blockIdx.x * 128;

    // staging addresses: each thread owns 32 contiguous bytes of one tile row
    const int srow  = tid >> 1;
    const int shalf = (tid & 1) * 16;
    const u16* gA = A  + (size_t)(m0 + srow) * K + shalf;
    const u16* gB = Bt + (size_t)(n0 + srow) * K + shalf;

    auto stage = [&](int buf, int k0) {
        u16* la = &As[buf][srow * 32 + shalf];
        u16* lb = &Bs[buf][srow * 32 + shalf];
#if USE_ASYNC_COPY
        __builtin_amdgcn_global_load_async_to_lds_b128(GLOBAL_AS(gA + k0), LDS_AS(la), 0, 0);
        __builtin_amdgcn_global_load_async_to_lds_b128(GLOBAL_AS(gA + k0), LDS_AS(la), 16, 0);
        __builtin_amdgcn_global_load_async_to_lds_b128(GLOBAL_AS(gB + k0), LDS_AS(lb), 0, 0);
        __builtin_amdgcn_global_load_async_to_lds_b128(GLOBAL_AS(gB + k0), LDS_AS(lb), 16, 0);
#else
        const uint4* sa = (const uint4*)(gA + k0);
        uint4*       da = (uint4*)la;
        da[0] = sa[0]; da[1] = sa[1];
        const uint4* sb = (const uint4*)(gB + k0);
        uint4*       db = (uint4*)lb;
        db[0] = sb[0]; db[1] = sb[1];
#endif
    };

    v8f acc[2][4] = {};

    stage(0, 0);
#if USE_ASYNC_COPY
    __builtin_amdgcn_s_wait_asynccnt(0);
#endif
    __syncthreads();

    int buf = 0;
    for (int k0 = 0; k0 < K; k0 += 32) {
        if (k0 + 32 < K) stage(buf ^ 1, k0 + 32);   // prefetch next tile (DMA)

        const u16* as = As[buf];
        const u16* bs = Bs[buf];
        Frag a[2];
#pragma unroll
        for (int mt = 0; mt < 2; ++mt) {
            int r = wm * 32 + mt * 16 + lrow;
            // A 16x32 layout: lanes 0-15 hold K 0-7 & 16-23, lanes 16-31 K 8-15 & 24-31
            a[mt].q[0] = *(const uint4*)(as + r * 32 + lhalf * 8);
            a[mt].q[1] = *(const uint4*)(as + r * 32 + 16 + lhalf * 8);
        }
#pragma unroll
        for (int nt = 0; nt < 4; ++nt) {
            Frag b;
            int c = wn * 64 + nt * 16 + lrow;
            // B 32x16 layout: lane = column, lanes 0-15 K 0-7 & 16-23, lanes 16-31 K 8-15 & 24-31
            b.q[0] = *(const uint4*)(bs + c * 32 + lhalf * 8);
            b.q[1] = *(const uint4*)(bs + c * 32 + 16 + lhalf * 8);
#pragma unroll
            for (int mt = 0; mt < 2; ++mt)
                acc[mt][nt] = __builtin_amdgcn_wmma_f32_16x16x32_bf16(
                    false, a[mt].v, false, b.v, (short)0, acc[mt][nt], false, false);
        }

#if USE_ASYNC_COPY
        __builtin_amdgcn_s_wait_asynccnt(0);        // next tile's DMA landed
#endif
        __syncthreads();                            // everyone done reading buf
        buf ^= 1;
    }

    // epilogue: C-layout = lane holds one column, 8 rows across the v8f
#pragma unroll
    for (int nt = 0; nt < 4; ++nt) {
        int col  = n0 + wn * 64 + nt * 16 + lrow;
        float bv = bias[col];
#pragma unroll
        for (int mt = 0; mt < 2; ++mt) {
            int rbase = m0 + wm * 32 + mt * 16 + lhalf * 8;
#pragma unroll
            for (int r = 0; r < 8; ++r) {
                float v = acc[mt][nt][r] + bv;
                if constexpr (OUT_BF16)
                    ((u16*)Cout)[(size_t)(rbase + r) * N + col] = f2bf(v);
                else
                    ((float*)Cout)[(size_t)(rbase + r) * N + col] = v;
            }
        }
    }
}

// ---------------------------------------------------------------------------
// Flash attention. qkv: bf16 [8192][3072] (q | k | v). attn: bf16 [8192][1024].
// Block = (b, h, 128 q-rows), 8 waves, each wave owns 16 q-rows.
// Per 64-key tile per wave: 8 WMMA (Q Kt) + 8 WMMA (P V). Row statistics
// live in half-wave lanes per the 16x16 C-layout; reductions are 16-lane
// shfl_xor butterflies. P reshapes C-layout -> A-fragment via a per-wave
// LDS tile; V is transposed into LDS once per key tile for the whole block.
// ---------------------------------------------------------------------------
__global__ __launch_bounds__(256) void mha_flash_kernel(const u16* __restrict__ qkv,
                                                        u16* __restrict__ attn) {
    constexpr int LDQ = 3072;
    constexpr int SEQ = 1024;
    __shared__ u16 Vt[64 * 64];       // [d][k] transposed V tile (8KB)
    __shared__ u16 Pl[8 * 16 * 64];   // per-wave P tiles       (16KB)

    const int tid   = threadIdx.x;
    const int lane  = tid & 31;
    const int wave  = tid >> 5;
    const int lrow  = lane & 15;
    const int lhalf = lane >> 4;

    const int bh    = blockIdx.x;
    const int b     = bh >> 4;
    const int h     = bh & 15;
    const int rowb  = b * SEQ;
    const int qrow0 = blockIdx.y * 128 + wave * 16;

    // Q fragments (16 rows x 64 d) held in registers for the whole pass
    Frag qf[2];
    {
        const u16* qb = qkv + (size_t)(rowb + qrow0 + lrow) * LDQ + h * 64;
#pragma unroll
        for (int kd = 0; kd < 2; ++kd) {
            qf[kd].q[0] = *(const uint4*)(qb + kd * 32 + lhalf * 8);
            qf[kd].q[1] = *(const uint4*)(qb + kd * 32 + 16 + lhalf * 8);
        }
    }

    float mrow[8], lsum[8];
    v8f   o[4] = {};
#pragma unroll
    for (int r = 0; r < 8; ++r) { mrow[r] = -3.402823466e38f; lsum[r] = 0.f; }

    u16* Pw = Pl + wave * (16 * 64);

    for (int kt0 = 0; kt0 < SEQ; kt0 += 64) {
        // ---- stage V^T tile: Vt[d][k] <- V[kt0+k][d], packed b32 stores ----
        __syncthreads();
        for (int idx = tid; idx < 64 * 32; idx += 256) {
            int d  = idx >> 5;
            int kp = idx & 31;
            const u16* vb = qkv + (size_t)(rowb + kt0 + kp * 2) * LDQ + 2048 + h * 64 + d;
            unsigned int lo = vb[0];
            unsigned int hi = vb[LDQ];
            *(unsigned int*)(Vt + d * 64 + kp * 2) = lo | (hi << 16);
        }
        __syncthreads();

        // ---- S = (Q K^T) * 1/sqrt(64) for 16x64 tile ----
        v8f s[4];
#pragma unroll
        for (int nt = 0; nt < 4; ++nt) {
            const u16* kb = qkv + (size_t)(rowb + kt0 + nt * 16 + lrow) * LDQ + 1024 + h * 64;
            v8f acc = {};
#pragma unroll
            for (int kd = 0; kd < 2; ++kd) {
                Frag kf;  // d 0-7 & 16-23 (lanes 0-15), 8-15 & 24-31 (lanes 16-31), matching qf
                kf.q[0] = *(const uint4*)(kb + kd * 32 + lhalf * 8);
                kf.q[1] = *(const uint4*)(kb + kd * 32 + 16 + lhalf * 8);
                acc = __builtin_amdgcn_wmma_f32_16x16x32_bf16(
                    false, qf[kd].v, false, kf.v, (short)0, acc, false, false);
            }
            s[nt] = acc * 0.125f;
        }

        // ---- online softmax update ----
        float alpha[8];
#pragma unroll
        for (int r = 0; r < 8; ++r) {
            float t = fmaxf(fmaxf(s[0][r], s[1][r]), fmaxf(s[2][r], s[3][r]));
#pragma unroll
            for (int off = 8; off > 0; off >>= 1) t = fmaxf(t, __shfl_xor(t, off, 32));
            float mnew = fmaxf(mrow[r], t);
            alpha[r] = __expf(mrow[r] - mnew);
            mrow[r]  = mnew;
        }
        float rs[8];
#pragma unroll
        for (int r = 0; r < 8; ++r) rs[r] = 0.f;
#pragma unroll
        for (int nt = 0; nt < 4; ++nt)
#pragma unroll
            for (int r = 0; r < 8; ++r) {
                float p = __expf(s[nt][r] - mrow[r]);
                s[nt][r] = p;
                rs[r] += p;
            }
#pragma unroll
        for (int r = 0; r < 8; ++r) {
            float t = rs[r];
#pragma unroll
            for (int off = 8; off > 0; off >>= 1) t += __shfl_xor(t, off, 32);
            lsum[r] = lsum[r] * alpha[r] + t;
#pragma unroll
            for (int dt = 0; dt < 4; ++dt) o[dt][r] *= alpha[r];
        }

        // ---- P: C-layout -> LDS -> A-fragment layout ----
#pragma unroll
        for (int nt = 0; nt < 4; ++nt)
#pragma unroll
            for (int r = 0; r < 8; ++r)
                Pw[(lhalf * 8 + r) * 64 + nt * 16 + lrow] = f2bf(s[nt][r]);
        __syncthreads();

        // ---- O += P V ----
#pragma unroll
        for (int kd = 0; kd < 2; ++kd) {
            Frag pf;
            pf.q[0] = *(const uint4*)(Pw + lrow * 64 + kd * 32 + lhalf * 8);
            pf.q[1] = *(const uint4*)(Pw + lrow * 64 + kd * 32 + 16 + lhalf * 8);
#pragma unroll
            for (int dt = 0; dt < 4; ++dt) {
                Frag vf;
                const u16* vb = Vt + (dt * 16 + lrow) * 64 + kd * 32 + lhalf * 8;
                vf.q[0] = *(const uint4*)vb;
                vf.q[1] = *(const uint4*)(vb + 16);
                o[dt] = __builtin_amdgcn_wmma_f32_16x16x32_bf16(
                    false, pf.v, false, vf.v, (short)0, o[dt], false, false);
            }
        }
    }

    // ---- normalize, store bf16 attention output [8192][1024] ----
#pragma unroll
    for (int r = 0; r < 8; ++r) {
        float inv = 1.f / lsum[r];
        int row = rowb + qrow0 + lhalf * 8 + r;
#pragma unroll
        for (int dt = 0; dt < 4; ++dt) {
            int col = h * 64 + dt * 16 + lrow;
            attn[(size_t)row * 1024 + col] = f2bf(o[dt][r] * inv);
        }
    }
}

// ---------------------------------------------------------------------------
// Launcher. Workspace layout (bytes):
//   [0,16M)   query bf16            [8192][1024]
//   [16M,22M) w_qkv^T bf16          [3072][1024]
//   [22M,24M) w_o^T bf16            [1024][1024]
//   [24M,72M) qkv bf16              [8192][3072]
//   [72M,88M) attention output bf16 [8192][1024]
// ---------------------------------------------------------------------------
extern "C" void kernel_launch(void* const* d_in, const int* in_sizes, int n_in,
                              void* d_out, int out_size, void* d_ws, size_t ws_size,
                              hipStream_t stream) {
    const float* query = (const float*)d_in[0];
    const float* w_qkv = (const float*)d_in[1];
    const float* b_qkv = (const float*)d_in[2];
    const float* w_o   = (const float*)d_in[3];
    const float* b_o   = (const float*)d_in[4];
    float*       out   = (float*)d_out;

    char* ws = (char*)d_ws;
    const size_t MB = 1024ull * 1024ull;
    u16* qbf   = (u16*)(ws);
    u16* wqkvT = (u16*)(ws + 16 * MB);
    u16* woT   = (u16*)(ws + 22 * MB);
    u16* qkvb  = (u16*)(ws + 24 * MB);
    u16* attnb = (u16*)(ws + 72 * MB);

    const int M = 8 * 1024;       // B*S
    const int K = 1024;           // hidden
    const int N1 = 3 * 1024;      // qkv
    const int N2 = 1024;          // out proj

    to_bf16_kernel<<<(M * K + 255) / 256, 256, 0, stream>>>(query, qbf, M * K);
    transpose_to_bf16_kernel<<<((size_t)K * N1 + 255) / 256, 256, 0, stream>>>(w_qkv, wqkvT, K, N1);
    transpose_to_bf16_kernel<<<((size_t)K * N2 + 255) / 256, 256, 0, stream>>>(w_o, woT, K, N2);

    gemm_bf16_kernel<true><<<dim3(N1 / 128, M / 128), 256, 0, stream>>>(
        qbf, wqkvT, b_qkv, qkvb, M, N1, K);

    mha_flash_kernel<<<dim3(8 * 16, 1024 / 128), 256, 0, stream>>>(qkvb, attnb);

    gemm_bf16_kernel<false><<<dim3(8, M / 128), 256, 0, stream>>>(
        attnb, woT, b_o, out, M, N2, K);
}
